// StackedMoEModel_90735479095890
// MI455X (gfx1250) — matrix operand order repaired and checked by reading, not compiled, not run on hardware
//
#include <hip/hip_runtime.h>

#define B_TOK 4096
#define DIN   1024
#define HDIM  1024
#define NEXP  8
#define EMBD  128
#define EHU   2048
#define EH2   1024
#define NSTEPS 4
#define NCLS  1000

typedef __attribute__((ext_vector_type(16))) __bf16 v16bf;
typedef __attribute__((ext_vector_type(4)))  __bf16 v4bf;
typedef __attribute__((ext_vector_type(8)))  float  v8f;
typedef __attribute__((ext_vector_type(4)))  int    int4v;

union FragBF { v16bf v; int4v u[2]; };

__device__ __forceinline__ float wred(float x) {
  #pragma unroll
  for (int m = 16; m >= 1; m >>= 1) x += __shfl_xor(x, m, 32);
  return x;
}

// ---------------------------------------------------------------------------
// elementwise fp32 -> bf16
__global__ __launch_bounds__(256)
void cvt_k(const float* __restrict__ in, __bf16* __restrict__ out, long long n4)
{
  long long i = (long long)blockIdx.x * 256 + threadIdx.x;
  if (i >= n4) return;
  float4 v = ((const float4*)in)[i];
  v4bf o;
  o[0] = (__bf16)v.x; o[1] = (__bf16)v.y; o[2] = (__bf16)v.z; o[3] = (__bf16)v.w;
  ((v4bf*)out)[i] = o;
}

// transpose + convert: in[R][C] fp32 -> out[C][R] bf16  (batched over z)
__global__ __launch_bounds__(256)
void tcvt_k(const float* __restrict__ in, __bf16* __restrict__ out,
            int R, int C, long long inZ, long long outZ)
{
  __shared__ float t[32][33];
  in  += (long long)blockIdx.z * inZ;
  out += (long long)blockIdx.z * outZ;
  int r0 = blockIdx.y * 32, c0 = blockIdx.x * 32;
  int tx = threadIdx.x & 31, ty = threadIdx.x >> 5;
  #pragma unroll
  for (int i = 0; i < 4; ++i) {
    int r = r0 + ty + i * 8;
    float v = 0.f;
    if (r < R && (c0 + tx) < C) v = in[(long long)r * C + c0 + tx];
    t[ty + i * 8][tx] = v;
  }
  __syncthreads();
  #pragma unroll
  for (int i = 0; i < 4; ++i) {
    int cc = c0 + ty + i * 8;
    int rr = r0 + tx;
    if (cc < C && rr < R) out[(long long)cc * R + rr] = (__bf16)t[tx][ty + i * 8];
  }
}

// ---------------------------------------------------------------------------
// LayerNorm over H=1024, one wave32 per row, bf16 output
__global__ __launch_bounds__(256)
void ln_k(const float* __restrict__ X, const float* __restrict__ g,
          const float* __restrict__ bta, __bf16* __restrict__ out)
{
  int wave = threadIdx.x >> 5, lane = threadIdx.x & 31;
  long long row = (long long)blockIdx.x * 8 + wave;
  const float* xr = X + row * 1024;
  float4 v[8]; float s = 0.f;
  #pragma unroll
  for (int i = 0; i < 8; ++i) {
    v[i] = *(const float4*)(xr + i * 128 + lane * 4);
    s += v[i].x + v[i].y + v[i].z + v[i].w;
  }
  s = wred(s);
  float mean = s * (1.f / 1024.f);
  float var = 0.f;
  #pragma unroll
  for (int i = 0; i < 8; ++i) {
    float dx = v[i].x - mean, dy = v[i].y - mean, dz = v[i].z - mean, dw = v[i].w - mean;
    var += dx * dx + dy * dy + dz * dz + dw * dw;
  }
  var = wred(var) * (1.f / 1024.f);
  float rs = rsqrtf(var + 1e-6f);
  #pragma unroll
  for (int i = 0; i < 8; ++i) {
    int c = i * 128 + lane * 4;
    float4 gg = *(const float4*)(g + c);
    float4 bb = *(const float4*)(bta + c);
    v4bf o;
    o[0] = (__bf16)((v[i].x - mean) * rs * gg.x + bb.x);
    o[1] = (__bf16)((v[i].y - mean) * rs * gg.y + bb.y);
    o[2] = (__bf16)((v[i].z - mean) * rs * gg.z + bb.z);
    o[3] = (__bf16)((v[i].w - mean) * rs * gg.w + bb.w);
    *(v4bf*)(out + row * 1024 + c) = o;
  }
}

// ---------------------------------------------------------------------------
// Router: distances to 8 embeddings, softmax probs, top-2 gates, expert lists
__global__ __launch_bounds__(256)
void router_k(const float* __restrict__ qbase, const float* __restrict__ emb,
              float* __restrict__ pbase, int* __restrict__ cnt,
              int* __restrict__ tokl, int* __restrict__ kls, float* __restrict__ gls)
{
  int wave = threadIdx.x >> 5, lane = threadIdx.x & 31;
  int tok = blockIdx.x * 8 + wave;
  const float* q = qbase + (long long)tok * (NSTEPS * EMBD);
  float4 qq = *(const float4*)(q + lane * 4);
  float sc[8];
  #pragma unroll
  for (int e = 0; e < 8; ++e) {
    float4 ev = *(const float4*)(emb + e * 128 + lane * 4);
    float dx = qq.x - ev.x, dy = qq.y - ev.y, dz = qq.z - ev.z, dw = qq.w - ev.w;
    sc[e] = -wred(dx * dx + dy * dy + dz * dz + dw * dw);
  }
  float mx = sc[0];
  #pragma unroll
  for (int e = 1; e < 8; ++e) mx = fmaxf(mx, sc[e]);
  float pe[8], se = 0.f;
  #pragma unroll
  for (int e = 0; e < 8; ++e) { pe[e] = __expf(sc[e] - mx); se += pe[e]; }
  float inv = 1.f / se;
  if (lane < 8) pbase[(long long)tok * (NSTEPS * NEXP) + lane] = pe[lane] * inv;
  int i1 = 0;
  #pragma unroll
  for (int e = 1; e < 8; ++e) if (sc[e] > sc[i1]) i1 = e;
  int i2 = (i1 == 0) ? 1 : 0;
  #pragma unroll
  for (int e = 0; e < 8; ++e) if (e != i1 && sc[e] > sc[i2]) i2 = e;
  float g2 = __expf(sc[i2] - sc[i1]);
  float den = 1.f + g2;
  if (lane == 0) {
    int s1 = atomicAdd(cnt + i1, 1);
    tokl[i1 * B_TOK + s1] = tok; kls[i1 * B_TOK + s1] = 0; gls[i1 * B_TOK + s1] = 1.f / den;
    int s2 = atomicAdd(cnt + i2, 1);
    tokl[i2 * B_TOK + s2] = tok; kls[i2 * B_TOK + s2] = 1; gls[i2 * B_TOK + s2] = g2 / den;
  }
}

// ---------------------------------------------------------------------------
// cur += sel[tok][0] + sel[tok][1]   (deterministic combine)
__global__ __launch_bounds__(256)
void combine_k(float* __restrict__ cur, const float* __restrict__ sel)
{
  long long i = ((long long)blockIdx.x * 256 + threadIdx.x) * 4;
  int tok = (int)(i >> 10);
  int c = (int)(i & 1023);
  float4 a = *(const float4*)(sel + (long long)tok * 2048 + c);
  float4 b = *(const float4*)(sel + (long long)tok * 2048 + 1024 + c);
  float4 cu = *(const float4*)(cur + i);
  cu.x += a.x + b.x; cu.y += a.y + b.y; cu.z += a.z + b.z; cu.w += a.w + b.w;
  *(float4*)(cur + i) = cu;
}

// ---------------------------------------------------------------------------
// Tiled bf16 WMMA GEMM with double-buffered async global->LDS staging.
// D[M,N] = A[M,K] * B[K,N] (+bias). Bt is B transposed: [N][K] bf16.
// EPI 0: fp32 store (+bias).  EPI 1: relu -> bf16 store (+bias).
// EPI 2: per-row gate scale, scatter to sel[(tok*2+k)][col] (+bias).
// EPI 3: fp32 store masked to col < maskN (+bias).
#define MT 128
#define NT 128
#define KT 32
#define KP 40                  // padded LDS row (elements) to dodge bank conflicts
#define ABUFB (MT * KP * 2)    // bytes per A staging buffer
#define BBUFB (NT * KP * 2)    // bytes per B staging buffer

#define ASYNC_B128(ldsaddr, gaddr) \
  asm volatile("global_load_async_to_lds_b128 %0, %1, off" \
               :: "v"(ldsaddr), "v"(gaddr) : "memory")

template<int EPI, bool GATHER>
__global__ __launch_bounds__(256)
void gemm_k(const __bf16* __restrict__ A, long long Az, int ldA,
            const __bf16* __restrict__ Bt, long long Bz,
            const float* __restrict__ bias, long long biasZ,
            char* __restrict__ Dp, long long Dz, int ldD,
            int Mfix, const int* __restrict__ cnt, int K,
            const int* __restrict__ tokl, const int* __restrict__ kls,
            const float* __restrict__ gls, int listZ, int maskN)
{
  __shared__ __bf16 As[2][MT * KP];
  __shared__ __bf16 Bs[2][NT * KP];
  __shared__ int idxs[MT];

  const int e = blockIdx.z;
  A  += (long long)e * Az;
  Bt += (long long)e * Bz;
  const float* bp2 = bias + (long long)e * biasZ;
  char* D = Dp + (long long)e * Dz;
  const int M = cnt ? cnt[e] : Mfix;
  const int m0 = blockIdx.y * MT;
  if (m0 >= M) return;
  const int n0 = blockIdx.x * NT;
  const int tid = threadIdx.x;
  const int lo = listZ * e;

  if (GATHER) {
    if (tid < MT) {
      int s = m0 + tid;
      idxs[tid] = (s < M) ? tokl[lo + s] : 0;
    }
    __syncthreads();
  }

  const int lrow = tid >> 1;
  const int lcol = (tid & 1) * 16;
  long long arow = GATHER ? (long long)idxs[lrow] : (long long)(m0 + lrow);
  const __bf16* ap  = A  + arow * (long long)ldA + lcol;
  const __bf16* bgp = Bt + (long long)(n0 + lrow) * K + lcol;

  // per-thread staging slot addresses (LDS byte offsets + global byte addrs)
  const unsigned lA0 = (unsigned)(size_t)(&As[0][0]) + (unsigned)((lrow * KP + lcol) * 2);
  const unsigned lB0 = (unsigned)(size_t)(&Bs[0][0]) + (unsigned)((lrow * KP + lcol) * 2);
  const unsigned long long gA0 = (unsigned long long)(size_t)ap;
  const unsigned long long gB0 = (unsigned long long)(size_t)bgp;

  const int wave = tid >> 5;
  const int lane = tid & 31;
  const int lm = lane & 15;
  const int hi = lane >> 4;
  const int wm = (wave & 3) * 32;   // wave M offset within tile
  const int wn = (wave >> 2) * 64;  // wave N offset within tile

  v8f acc[2][4];
  #pragma unroll
  for (int mi = 0; mi < 2; ++mi)
    #pragma unroll
    for (int ni = 0; ni < 4; ++ni)
      #pragma unroll
      for (int r = 0; r < 8; ++r) acc[mi][ni][r] = 0.f;

  auto stage = [&](int b, int kk) {
    unsigned la  = lA0 + (unsigned)(b * ABUFB);
    unsigned lb  = lB0 + (unsigned)(b * BBUFB);
    unsigned long long ga = gA0 + (unsigned long long)kk * 2;
    unsigned long long gb = gB0 + (unsigned long long)kk * 2;
    ASYNC_B128(la,        ga);
    ASYNC_B128(la + 16u,  ga + 16ull);
    ASYNC_B128(lb,        gb);
    ASYNC_B128(lb + 16u,  gb + 16ull);
  };

  auto compute = [&](int b) {
    const __bf16* Ab = &As[b][0];
    const __bf16* Bb = &Bs[b][0];
    FragBF af[2], bf2[4];
    #pragma unroll
    for (int mi = 0; mi < 2; ++mi) {
      // A 16x32 bf16 layout: lane(m,hi): elems 0..7 = K[8hi..], 8..15 = K[16+8hi..]
      const __bf16* p2 = Ab + (wm + 16 * mi + lm) * KP + 8 * hi;
      af[mi].u[0] = *(const int4v*)p2;
      af[mi].u[1] = *(const int4v*)(p2 + 16);
    }
    #pragma unroll
    for (int ni = 0; ni < 4; ++ni) {
      // B 32x16 bf16 layout: lane(n,hi): elems 0..7 = K[8hi..], 8..15 = K[16+8hi..]
      const __bf16* p2 = Bb + (wn + 16 * ni + lm) * KP + 8 * hi;
      bf2[ni].u[0] = *(const int4v*)p2;
      bf2[ni].u[1] = *(const int4v*)(p2 + 16);
    }
    #pragma unroll
    for (int mi = 0; mi < 2; ++mi)
      #pragma unroll
      for (int ni = 0; ni < 4; ++ni)
        acc[mi][ni] = __builtin_amdgcn_wmma_f32_16x16x32_bf16(
            false, af[mi].v, false, bf2[ni].v, (short)0, acc[mi][ni], false, false);
  };

  // Double-buffered pipeline; K is a multiple of 2*KT (=64) for all our shapes.
  stage(0, 0);
  for (int k0 = 0; k0 < K; k0 += 2 * KT) {
    stage(1, k0 + KT);                                  // prefetch buf1
    asm volatile("s_wait_asynccnt 0x4" ::: "memory");   // buf0 group landed
    __syncthreads();
    compute(0);
    __syncthreads();                                    // all waves done reading buf0
    int nk = (k0 + 2 * KT < K) ? (k0 + 2 * KT) : 0;     // tail: harmless dummy reload
    stage(0, nk);                                       // prefetch buf0
    asm volatile("s_wait_asynccnt 0x4" ::: "memory");   // buf1 group landed
    __syncthreads();
    compute(1);
    __syncthreads();                                    // all waves done reading buf1
  }
  // dangling prefetch group is retired by s_endpgm's implicit wait-idle

  // epilogue: acc elem r of lane (n=lm, hi) -> row (r + 8*hi), col n
  #pragma unroll
  for (int mi = 0; mi < 2; ++mi) {
    const int row0 = m0 + wm + 16 * mi + 8 * hi;
    #pragma unroll
    for (int ni = 0; ni < 4; ++ni) {
      const int col = n0 + wn + 16 * ni + lm;
      float bv = (EPI == 3) ? ((col < maskN) ? bp2[col] : 0.f) : bp2[col];
      #pragma unroll
      for (int r = 0; r < 8; ++r) {
        const int row = row0 + r;
        if (row >= M) break;
        float val = acc[mi][ni][r] + bv;
        if (EPI == 0) {
          ((float*)D)[(long long)row * ldD + col] = val;
        } else if (EPI == 1) {
          ((__bf16*)D)[(long long)row * ldD + col] = (__bf16)fmaxf(val, 0.f);
        } else if (EPI == 2) {
          int t = tokl[lo + row];
          int kk = kls[lo + row];
          float gv = gls[lo + row];
          ((float*)D)[((long long)t * 2 + kk) * ldD + col] = gv * val;
        } else {
          if (col < maskN) ((float*)D)[(long long)row * ldD + col] = val;
        }
      }
    }
  }
}

// ---------------------------------------------------------------------------
extern "C" void kernel_launch(void* const* d_in, const int* in_sizes, int n_in,
                              void* d_out, int out_size, void* d_ws, size_t ws_size,
                              hipStream_t stream)
{
  (void)in_sizes; (void)n_in; (void)out_size;
  const float* x    = (const float*)d_in[0];
  const float* Wp   = (const float*)d_in[1];
  const float* bp   = (const float*)d_in[2];
  const float* ln1g = (const float*)d_in[3];
  const float* ln1b = (const float*)d_in[4];
  const float* Wr   = (const float*)d_in[5];
  const float* br   = (const float*)d_in[6];
  const float* emb  = (const float*)d_in[7];
  const float* W1   = (const float*)d_in[8];
  const float* b1   = (const float*)d_in[9];
  const float* W2   = (const float*)d_in[10];
  const float* b2   = (const float*)d_in[11];
  const float* W3   = (const float*)d_in[12];
  const float* b3   = (const float*)d_in[13];
  const float* ln2g = (const float*)d_in[14];
  const float* ln2b = (const float*)d_in[15];
  const float* Wc   = (const float*)d_in[16];
  const float* bc   = (const float*)d_in[17];

  char* p = (char*)d_ws;
  auto take = [&](long long bytes) -> char* {
    char* r = p; p += (bytes + 255) & ~255LL; return r;
  };
  __bf16* xb   = (__bf16*)take((long long)B_TOK * DIN * 2);
  __bf16* WpT  = (__bf16*)take((long long)DIN * HDIM * 2);
  __bf16* WrT  = (__bf16*)take((long long)HDIM * EMBD * 2);
  __bf16* WcT  = (__bf16*)take(1024LL * HDIM * 2);               // padded N=1024
  __bf16* W1T  = (__bf16*)take((long long)NEXP * HDIM * EHU * 2);
  __bf16* W2T  = (__bf16*)take((long long)NEXP * EHU * EH2 * 2);
  __bf16* W3T  = (__bf16*)take((long long)NEXP * EH2 * HDIM * 2);
  float*  cur  = (float*)take((long long)B_TOK * HDIM * 4);
  __bf16* nrm  = (__bf16*)take((long long)B_TOK * HDIM * 2);
  __bf16* h1   = (__bf16*)take((long long)NEXP * B_TOK * EHU * 2);
  __bf16* h2   = (__bf16*)take((long long)NEXP * B_TOK * EH2 * 2);
  float*  sel  = (float*)take((long long)B_TOK * 2 * HDIM * 4);
  int*    cnt  = (int*)take(NEXP * 4);
  int*    tokl = (int*)take((long long)NEXP * B_TOK * 4);
  int*    kls  = (int*)take((long long)NEXP * B_TOK * 4);
  float*  gls  = (float*)take((long long)NEXP * B_TOK * 4);
  if ((size_t)(p - (char*)d_ws) > ws_size) return;

  float* out   = (float*)d_out;
  float* out_q = out + (long long)B_TOK * NCLS;
  float* out_p = out_q + (long long)B_TOK * NSTEPS * EMBD;

  // --- one-time conversions (weights -> bf16, transposed [N][K]) ---
  cvt_k<<<dim3((B_TOK * DIN / 4 + 255) / 256), 256, 0, stream>>>(x, xb, (long long)B_TOK * DIN / 4);
  tcvt_k<<<dim3(HDIM / 32, DIN / 32, 1), 256, 0, stream>>>(Wp, WpT, DIN, HDIM, 0, 0);
  tcvt_k<<<dim3(EMBD / 32, HDIM / 32, 1), 256, 0, stream>>>(Wr, WrT, HDIM, EMBD, 0, 0);
  tcvt_k<<<dim3((NCLS + 31) / 32, HDIM / 32, 1), 256, 0, stream>>>(Wc, WcT, HDIM, NCLS, 0, 0);
  tcvt_k<<<dim3(EHU / 32, HDIM / 32, NEXP), 256, 0, stream>>>(W1, W1T, HDIM, EHU,
      (long long)HDIM * EHU, (long long)HDIM * EHU);
  tcvt_k<<<dim3(EH2 / 32, EHU / 32, NEXP), 256, 0, stream>>>(W2, W2T, EHU, EH2,
      (long long)EHU * EH2, (long long)EHU * EH2);
  tcvt_k<<<dim3(HDIM / 32, EH2 / 32, NEXP), 256, 0, stream>>>(W3, W3T, EH2, HDIM,
      (long long)EH2 * HDIM, (long long)EH2 * HDIM);

  // cur = x @ Wp + bp
  gemm_k<0, false><<<dim3(HDIM / NT, B_TOK / MT, 1), 256, 0, stream>>>(
      xb, 0, DIN, WpT, 0, bp, 0, (char*)cur, 0, HDIM,
      B_TOK, nullptr, DIN, nullptr, nullptr, nullptr, 0, 0);

  for (int s = 0; s < NSTEPS; ++s) {
    ln_k<<<dim3(B_TOK / 8), 256, 0, stream>>>(cur, ln1g, ln1b, nrm);
    // q = norm @ Wr + br  -> written straight into all_q[:, s, :]
    gemm_k<0, false><<<dim3(1, B_TOK / MT, 1), 256, 0, stream>>>(
        nrm, 0, HDIM, WrT, 0, br, 0, (char*)(out_q + s * EMBD), 0, NSTEPS * EMBD,
        B_TOK, nullptr, HDIM, nullptr, nullptr, nullptr, 0, 0);
    (void)hipMemsetAsync(cnt, 0, NEXP * sizeof(int), stream);
    router_k<<<dim3(B_TOK / 8), 256, 0, stream>>>(out_q + s * EMBD, emb,
        out_p + s * NEXP, cnt, tokl, kls, gls);
    // h1 = relu(gather(norm) @ W1[e] + b1[e])
    gemm_k<1, true><<<dim3(EHU / NT, B_TOK / MT, NEXP), 256, 0, stream>>>(
        nrm, 0, HDIM, W1T, (long long)HDIM * EHU, b1, EHU,
        (char*)h1, (long long)B_TOK * EHU * 2, EHU,
        0, cnt, HDIM, tokl, nullptr, nullptr, B_TOK, 0);
    // h2 = relu(h1 @ W2[e] + b2[e])
    gemm_k<1, false><<<dim3(EH2 / NT, B_TOK / MT, NEXP), 256, 0, stream>>>(
        h1, (long long)B_TOK * EHU, EHU, W2T, (long long)EHU * EH2, b2, EH2,
        (char*)h2, (long long)B_TOK * EH2 * 2, EH2,
        0, cnt, EHU, nullptr, nullptr, nullptr, 0, 0);
    // sel[tok][k] = gate * (h2 @ W3[e] + b3[e])
    gemm_k<2, false><<<dim3(HDIM / NT, B_TOK / MT, NEXP), 256, 0, stream>>>(
        h2, (long long)B_TOK * EH2, EH2, W3T, (long long)EH2 * HDIM, b3, HDIM,
        (char*)sel, 0, HDIM,
        0, cnt, EH2, tokl, kls, gls, B_TOK, 0);
    combine_k<<<dim3(B_TOK * HDIM / 1024), 256, 0, stream>>>(cur, sel);
  }

  ln_k<<<dim3(B_TOK / 8), 256, 0, stream>>>(cur, ln2g, ln2b, nrm);
  // logits = norm2 @ Wc + bc  (N padded to 1024, masked store to 1000)
  gemm_k<3, false><<<dim3(1024 / NT, B_TOK / MT, 1), 256, 0, stream>>>(
      nrm, 0, HDIM, WcT, 0, bc, 0, (char*)out, 0, NCLS,
      B_TOK, nullptr, HDIM, nullptr, nullptr, nullptr, 0, NCLS);
}
